// ConditionalMixtureDiscreteMADE_84035330113875
// MI455X (gfx1250) — compile-verified, matrix operand order checked by reading, not run on hardware
//
#include <hip/hip_runtime.h>
#include <hip/hip_bf16.h>

typedef __attribute__((ext_vector_type(16))) __bf16          bf16x16;
typedef __attribute__((ext_vector_type(8)))  float           v8f;
typedef __attribute__((ext_vector_type(8)))  unsigned short  u16x8;
typedef __attribute__((ext_vector_type(4)))  unsigned short  u16x4;
typedef __attribute__((ext_vector_type(4)))  float           f32x4;

#define NI_C   627
#define NP_C   321024      // 512*627
#define PF_C   321024.0f

__device__ __forceinline__ unsigned short f2bf(float f) {
    unsigned int u = __float_as_uint(f);
    u += 0x7FFFu + ((u >> 16) & 1u);
    return (unsigned short)(u >> 16);
}

__device__ __forceinline__ v8f zero8() {
    v8f z;
#pragma unroll
    for (int i = 0; i < 8; ++i) z[i] = 0.0f;
    return z;
}

// Async 16-byte global -> LDS copy (CDNA5 GLOBAL_LOAD_ASYNC_TO_LDS_B128,
// tracked by ASYNCcnt). dst: LDS byte address (low 32 bits of flat addr).
__device__ __forceinline__ void async_copy_b128(void* lds_dst, const void* gsrc) {
    unsigned ldsoff = (unsigned)(uintptr_t)lds_dst;
    asm volatile("global_load_async_to_lds_b128 %0, %1, off"
                 :: "v"(ldsoff), "v"(gsrc) : "memory");
}
__device__ __forceinline__ void wait_async0() {
    asm volatile("s_wait_asynccnt 0x0" ::: "memory");
}

// Fragment layout (16-bit A/B, 16x16x32): lane L holds row/col (L&15);
// two contiguous 8-elem chunks at K = 8*(L>>4) and K = 16 + 8*(L>>4).
__device__ __forceinline__ bf16x16 load_frag(const unsigned short* base, int stride, int lane) {
    int r = lane & 15, h = lane >> 4;
    const unsigned short* p0 = base + r * stride + 8 * h;
    u16x8 c0 = *(const u16x8*)p0;
    u16x8 c1 = *(const u16x8*)(p0 + 16);
    union { u16x8 hh[2]; bf16x16 v; } u;
    u.hh[0] = c0; u.hh[1] = c1;
    return u.v;
}

// ---------------------------------------------------------------------------
// Weight packer: fp32 (Nact x Kact) -> bf16 (Npad x Kpad), optional MADE mask
// ---------------------------------------------------------------------------
__global__ void pack_w(const float* __restrict__ W, unsigned short* __restrict__ Wbf,
                       int Nact, int Kact, int Npad, int Kpad, int masked, int deg_off) {
    int idx = blockIdx.x * 256 + threadIdx.x;
    if (idx >= Npad * Kpad) return;
    int n = idx / Kpad, k = idx - n * Kpad;
    float v = 0.0f;
    if (n < Nact && k < Kact) {
        v = W[(size_t)n * Kact + k];
        if (masked) {
            int od = (n % NI_C) - deg_off, id = k % NI_C;
            if (od < id) v = 0.0f;
        }
    }
    Wbf[idx] = f2bf(v);
}

__global__ void zero_f(float* p, int n) {
    int i = blockIdx.x * 256 + threadIdx.x;
    if (i < n) p[i] = 0.0f;
}

// ---------------------------------------------------------------------------
// Generic WMMA GEMM:  C[M x N] = act( T(A) @ Wbf^T + bias )
//   a_mode 0: A plain fp32 (lda)
//   a_mode 1: A with BN: relu(A*scale[k] + shift[k])
//   a_mode 2: virtual conv1: relu(bn1(theta[p]*cw1[k] + cb1[k])), theta from outm
//   do_stats: atomically accumulate per-column sum / sumsq of C
// Block 256 (8 waves), tile 128x64, BK=32. M multiple of 128 (guaranteed here).
// B tile staged via GLOBAL_LOAD_ASYNC_TO_LDS_B128 (ASYNCcnt), A tile staged by
// VALU (fp32->bf16 transform) with a vectorized float4 path when lda%4==0.
// ---------------------------------------------------------------------------
__global__ void gemm_wmma(const float* __restrict__ A, int lda,
                          const unsigned short* __restrict__ Wbf, int Kpad, int Kact,
                          const float* __restrict__ bias, int Nact,
                          float* __restrict__ C, int ldc,
                          int a_mode, int act_relu, int do_stats,
                          const float* __restrict__ scale, const float* __restrict__ shift,
                          const float* __restrict__ cw1, const float* __restrict__ cb1,
                          const float* __restrict__ outm, int ldo,
                          float* __restrict__ gsum, float* __restrict__ gsq) {
    __shared__ __align__(32) unsigned short lA[128 * 40];
    __shared__ __align__(32) unsigned short lB[64 * 40];
    __shared__ float csum[64], csq[64];

    const int tid  = threadIdx.x;
    const int lane = tid & 31;
    const int wid  = tid >> 5;
    const int wm   = wid & 3;      // 4 waves along M
    const int wn   = wid >> 2;     // 2 waves along N
    const int row0 = blockIdx.x * 128;
    const int n0   = blockIdx.y * 64;
    const bool vecA = (a_mode != 2) && ((lda & 3) == 0);

    v8f acc[2][2];
#pragma unroll
    for (int mi = 0; mi < 2; ++mi)
#pragma unroll
        for (int ni = 0; ni < 2; ++ni) acc[mi][ni] = zero8();

    // B-tile async copy geometry: 64 rows x 32 cols bf16 = 4KB = 256 x 16B
    const int br = tid >> 2;            // row 0..63
    const int bc = (tid & 3) << 3;      // col 0,8,16,24

    const int nkt = Kpad >> 5;
    for (int kt = 0; kt < nkt; ++kt) {
        const int kbase = kt << 5;

        // ---- stage B: one async b128 per thread, global -> LDS ----
        async_copy_b128(&lB[br * 40 + bc],
                        Wbf + (size_t)(n0 + br) * Kpad + kbase + bc);

        // ---- stage A (fp32 -> transformed bf16), overlaps with async B ----
        if (vecA) {
            for (int idx = tid; idx < 128 * 8; idx += 256) {
                int r = idx >> 3, cc = (idx & 7) << 2;
                int k = kbase + cc;
                const float* src = &A[(size_t)(row0 + r) * lda + k];
                f32x4 x = {0.0f, 0.0f, 0.0f, 0.0f};
                if (k + 3 < Kact) {
                    x = *(const f32x4*)src;
                } else {
#pragma unroll
                    for (int j = 0; j < 4; ++j)
                        if (k + j < Kact) x[j] = src[j];
                }
                u16x4 o;
#pragma unroll
                for (int j = 0; j < 4; ++j) {
                    float v = x[j];
                    if (a_mode == 1) v = fmaxf(v * scale[k + j] + shift[k + j], 0.0f);
                    if (k + j >= Kact) v = 0.0f;
                    o[j] = f2bf(v);
                }
                *(u16x4*)&lA[r * 40 + cc] = o;
            }
        } else {
            for (int idx = tid; idx < 128 * 32; idx += 256) {
                int r = idx >> 5, c = idx & 31;
                int k = kbase + c;
                float v = 0.0f;
                if (k < Kact) {
                    if (a_mode == 0) {
                        v = A[(size_t)(row0 + r) * lda + k];
                    } else {
                        int p = row0 + r;
                        int b = p / NI_C, l = p - b * NI_C;
                        float th = outm[(size_t)b * ldo + NI_C + l];
                        v = fmaxf((th * cw1[k] + cb1[k]) * scale[k] + shift[k], 0.0f);
                    }
                }
                lA[r * 40 + c] = f2bf(v);
            }
        }

        wait_async0();      // this wave's async B-chunk landed in LDS
        __syncthreads();    // all waves' A + B staging visible

        bf16x16 a0 = load_frag(lA + (wm * 32) * 40,      40, lane);
        bf16x16 a1 = load_frag(lA + (wm * 32 + 16) * 40, 40, lane);
        bf16x16 b0 = load_frag(lB + (wn * 32) * 40,      40, lane);
        bf16x16 b1 = load_frag(lB + (wn * 32 + 16) * 40, 40, lane);
        acc[0][0] = __builtin_amdgcn_wmma_f32_16x16x32_bf16(false, a0, false, b0, (short)0, acc[0][0], false, false);
        acc[0][1] = __builtin_amdgcn_wmma_f32_16x16x32_bf16(false, a0, false, b1, (short)0, acc[0][1], false, false);
        acc[1][0] = __builtin_amdgcn_wmma_f32_16x16x32_bf16(false, a1, false, b0, (short)0, acc[1][0], false, false);
        acc[1][1] = __builtin_amdgcn_wmma_f32_16x16x32_bf16(false, a1, false, b1, (short)0, acc[1][1], false, false);
        __syncthreads();
    }

    // ---- epilogue: bias (+relu), store, optional column stats ----
    if (do_stats) {
        if (tid < 64) { csum[tid] = 0.0f; csq[tid] = 0.0f; }
    }
    __syncthreads();

    const int h = lane >> 4, cl = lane & 15;
#pragma unroll
    for (int mi = 0; mi < 2; ++mi) {
#pragma unroll
        for (int ni = 0; ni < 2; ++ni) {
            int colT = n0 + wn * 32 + ni * 16 + cl;
            float bval = (colT < Nact) ? bias[colT] : 0.0f;
            float ps = 0.0f, pq = 0.0f;
#pragma unroll
            for (int v = 0; v < 8; ++v) {
                int rowT = row0 + wm * 32 + mi * 16 + v + 8 * h;
                float val = acc[mi][ni][v] + bval;
                if (act_relu) val = fmaxf(val, 0.0f);
                if (colT < Nact) {
                    C[(size_t)rowT * ldc + colT] = val;
                    ps += val; pq += val * val;
                }
            }
            if (do_stats) {
                ps += __shfl_xor(ps, 16, 32);
                pq += __shfl_xor(pq, 16, 32);
                if (h == 0 && colT < Nact) {
                    atomicAdd(&csum[wn * 32 + ni * 16 + cl], ps);
                    atomicAdd(&csq[wn * 32 + ni * 16 + cl], pq);
                }
            }
        }
    }
    if (do_stats) {
        __syncthreads();
        if (tid < 64) {
            int c = n0 + tid;
            if (c < Nact) {
                atomicAdd(&gsum[c], csum[tid]);
                atomicAdd(&gsq[c],  csq[tid]);
            }
        }
    }
}

// ---------------------------------------------------------------------------
// Fused conv5 + log_softmax + NLL. Never materializes pred (512x300x627).
// Block 256 = 8 waves, 128 positions per block, each wave owns 16 rows.
// K = 180 (pad 192), N = 300 (19 n-tiles). Online logsumexp per row.
// ---------------------------------------------------------------------------
__global__ void conv5_nll(const float* __restrict__ X4,            // P x 192 fp32
                          const unsigned short* __restrict__ W5,   // 304 x 192 bf16
                          const float* __restrict__ cb5,           // 300
                          const float* __restrict__ sc4, const float* __restrict__ sh4,
                          const float* __restrict__ inputs,        // 512 x 627
                          float* __restrict__ nll) {               // P
    __shared__ __align__(32) unsigned short lA[128 * 208];
    const int tid = threadIdx.x, lane = tid & 31, w = tid >> 5;
    const int p0 = blockIdx.x * 128;

    for (int idx = tid; idx < 128 * 48; idx += 256) {
        int r = idx / 48, cc = (idx - r * 48) << 2;
        const float* src = &X4[(size_t)(p0 + r) * 192 + cc];
        f32x4 x = {0.0f, 0.0f, 0.0f, 0.0f};
        if (cc + 3 < 180) x = *(const f32x4*)src;
        u16x4 o;
#pragma unroll
        for (int j = 0; j < 4; ++j) {
            int c = cc + j;
            float v = (c < 180) ? fmaxf(x[j] * sc4[c] + sh4[c], 0.0f) : 0.0f;
            o[j] = f2bf(v);
        }
        *(u16x4*)&lA[r * 208 + cc] = o;
    }
    __syncthreads();

    const int h = lane >> 4, cl = lane & 15;
    bf16x16 af[6];
#pragma unroll
    for (int ks = 0; ks < 6; ++ks)
        af[ks] = load_frag(lA + (w * 16) * 208 + ks * 32, 208, lane);

    int   tv[8];
    float m[8], s[8], vt[8];
#pragma unroll
    for (int v = 0; v < 8; ++v) {
        int p = p0 + w * 16 + v + 8 * h;
        int b = p / NI_C, l = p - b * NI_C;
        tv[v] = (int)inputs[(size_t)b * NI_C + l];
        m[v] = -3.0e38f; s[v] = 0.0f; vt[v] = 0.0f;
    }

    for (int nt = 0; nt < 19; ++nt) {
        v8f acc = zero8();
#pragma unroll
        for (int ks = 0; ks < 6; ++ks) {
            bf16x16 bf = load_frag(W5 + (size_t)(nt * 16) * 192 + ks * 32, 192, lane);
            acc = __builtin_amdgcn_wmma_f32_16x16x32_bf16(false, af[ks], false, bf, (short)0, acc, false, false);
        }
        int col = nt * 16 + cl;
        bool valid = col < 300;
        float bv = valid ? cb5[col] : 0.0f;
#pragma unroll
        for (int v = 0; v < 8; ++v) {
            float raw = acc[v] + bv;
            // target logit extraction
            int tcol = tv[v] - nt * 16;
            float got = __shfl(raw, (h << 4) + (tcol & 15), 32);
            if (tcol >= 0 && tcol < 16) vt[v] = got;
            // online logsumexp across the 16 lanes holding this row
            float x = valid ? raw : -3.0e38f;
#pragma unroll
            for (int off = 1; off < 16; off <<= 1) x = fmaxf(x, __shfl_xor(x, off, 32));
            float nm = fmaxf(m[v], x);
            float e = valid ? __expf(raw - nm) : 0.0f;
#pragma unroll
            for (int off = 1; off < 16; off <<= 1) e += __shfl_xor(e, off, 32);
            s[v] = s[v] * __expf(m[v] - nm) + e;
            m[v] = nm;
        }
    }
#pragma unroll
    for (int v = 0; v < 8; ++v) {
        if (cl == 0) {
            float lse = m[v] + __logf(s[v]);
            nll[p0 + w * 16 + v + 8 * h] = lse - vt[v];
        }
    }
}

// ---------------------------------------------------------------------------
// Sum / sumsq of theta over all P positions (theta = outm[:, 627:1254])
// ---------------------------------------------------------------------------
__global__ void theta_stats(const float* __restrict__ outm, int ldo, float* __restrict__ S) {
    float s1 = 0.0f, s2 = 0.0f;
    for (int p = blockIdx.x * 256 + threadIdx.x; p < NP_C; p += gridDim.x * 256) {
        int b = p / NI_C, l = p - b * NI_C;
        float th = outm[(size_t)b * ldo + NI_C + l];
        s1 += th; s2 += th * th;
    }
    for (int off = 1; off < 32; off <<= 1) {
        s1 += __shfl_xor(s1, off, 32);
        s2 += __shfl_xor(s2, off, 32);
    }
    __shared__ float a1[8], a2[8];
    int lane = threadIdx.x & 31, w = threadIdx.x >> 5;
    if (lane == 0) { a1[w] = s1; a2[w] = s2; }
    __syncthreads();
    if (threadIdx.x == 0) {
        float t1 = 0.0f, t2 = 0.0f;
        for (int i = 0; i < 8; ++i) { t1 += a1[i]; t2 += a2[i]; }
        atomicAdd(&S[0], t1); atomicAdd(&S[1], t2);
    }
}

// BN1 scale/shift analytically from Sθ, Sθ² (conv1 has C_in=1)
__global__ void bn1_fin(const float* S, const float* cw1, const float* cb1,
                        const float* g, const float* be, float* sc, float* sh) {
    int c = threadIdx.x;
    if (c >= 90) return;
    float w = cw1[c], b = cb1[c];
    float sum = w * S[0] + PF_C * b;
    float sq  = w * w * S[1] + 2.0f * w * b * S[0] + PF_C * b * b;
    float mean = sum / PF_C;
    float var  = sq / PF_C - mean * mean;
    float rs = rsqrtf(var + 1e-5f);
    float scale = g[c] * rs;
    sc[c] = scale;
    sh[c] = be[c] - mean * scale;
}

__global__ void bn_fin(const float* sum, const float* sq, const float* g, const float* be,
                       float* sc, float* sh, int C) {
    int c = blockIdx.x * 256 + threadIdx.x;
    if (c >= C) return;
    float mean = sum[c] / PF_C;
    float var  = sq[c] / PF_C - mean * mean;
    float rs = rsqrtf(var + 1e-5f);
    float scale = g[c] * rs;
    sc[c] = scale;
    sh[c] = be[c] - mean * scale;
}

__global__ void nll_head_mean(const float* __restrict__ nll, float* __restrict__ mu) {
    int tid = threadIdx.x;
    float s = 0.0f;
    for (int i = tid; i < 1024; i += 256) {
        int b = i >> 1, l = i & 1;
        s += nll[b * NI_C + l];
    }
    for (int off = 1; off < 32; off <<= 1) s += __shfl_xor(s, off, 32);
    __shared__ float a[8];
    if ((tid & 31) == 0) a[tid >> 5] = s;
    __syncthreads();
    if (tid == 0) {
        float t = 0.0f;
        for (int i = 0; i < 8; ++i) t += a[i];
        mu[0] = t / 1024.0f;
    }
}

__global__ void final_ll(const float* __restrict__ outm, int ldo,
                         const float* __restrict__ inputs,
                         const float* __restrict__ nll,
                         const float* __restrict__ mu,
                         float* __restrict__ out) {
    int b = blockIdx.x, tid = threadIdx.x;
    float s = 0.0f;
    for (int j = tid; j < 625; j += 256) {
        int l = 2 + j;
        float graw = outm[(size_t)b * ldo + l];
        float g = 1.0f / (1.0f + __expf(-graw));
        float inp = inputs[(size_t)b * NI_C + l];
        float n = nll[b * NI_C + l];
        s += (inp > 0.0f) ? (0.1f * __logf(g + 1e-10f) - n)
                          : (0.1f * __logf(1.0f - g + 1e-10f));
    }
    for (int off = 1; off < 32; off <<= 1) s += __shfl_xor(s, off, 32);
    __shared__ float a[8];
    if ((tid & 31) == 0) a[tid >> 5] = s;
    __syncthreads();
    if (tid == 0) {
        float t = 0.0f;
        for (int i = 0; i < 8; ++i) t += a[i];
        out[b] = t - mu[0];
    }
}

// ---------------------------------------------------------------------------
extern "C" void kernel_launch(void* const* d_in, const int* in_sizes, int n_in,
                              void* d_out, int out_size, void* d_ws, size_t ws_size,
                              hipStream_t stream) {
    const float* inputs = (const float*)d_in[0];
    const float* W_in   = (const float*)d_in[1];
    const float* b_in   = (const float*)d_in[2];
    const float* W_h1   = (const float*)d_in[3];
    const float* b_h1   = (const float*)d_in[4];
    const float* W_h2   = (const float*)d_in[5];
    const float* b_h2   = (const float*)d_in[6];
    const float* W_out  = (const float*)d_in[7];
    const float* b_out  = (const float*)d_in[8];
    const float* cw1 = (const float*)d_in[9],  *cb1 = (const float*)d_in[10];
    const float* g1  = (const float*)d_in[11], *be1 = (const float*)d_in[12];
    const float* cw2 = (const float*)d_in[13], *cb2 = (const float*)d_in[14];
    const float* g2  = (const float*)d_in[15], *be2 = (const float*)d_in[16];
    const float* cw3 = (const float*)d_in[17], *cb3 = (const float*)d_in[18];
    const float* g3  = (const float*)d_in[19], *be3 = (const float*)d_in[20];
    const float* cw4 = (const float*)d_in[21], *cb4 = (const float*)d_in[22];
    const float* g4  = (const float*)d_in[23], *be4 = (const float*)d_in[24];
    const float* cw5 = (const float*)d_in[25], *cb5 = (const float*)d_in[26];

    const int P = NP_C;
    size_t off = 0;
    auto alloc = [&](size_t bytes) -> char* {
        off = (off + 255) & ~(size_t)255;
        char* p = (char*)d_ws + off;
        off += bytes;
        return p;
    };
    unsigned short* wbf_in  = (unsigned short*)alloc((size_t)1024 * 640  * 2);
    unsigned short* wbf_h1  = (unsigned short*)alloc((size_t)1024 * 1024 * 2);
    unsigned short* wbf_h2  = (unsigned short*)alloc((size_t)1024 * 1024 * 2);
    unsigned short* wbf_out = (unsigned short*)alloc((size_t)1280 * 1024 * 2);
    unsigned short* wbf_c2  = (unsigned short*)alloc((size_t)192 * 96  * 2);
    unsigned short* wbf_c3  = (unsigned short*)alloc((size_t)192 * 192 * 2);
    unsigned short* wbf_c4  = (unsigned short*)alloc((size_t)192 * 192 * 2);
    unsigned short* wbf_c5  = (unsigned short*)alloc((size_t)304 * 192 * 2);
    float* h1   = (float*)alloc((size_t)512 * 1024 * 4);
    float* h2   = (float*)alloc((size_t)512 * 1024 * 4);
    float* h3   = (float*)alloc((size_t)512 * 1024 * 4);
    float* outm = (float*)alloc((size_t)512 * 1280 * 4);
    float* X2   = (float*)alloc((size_t)P * 192 * 4);
    float* X3   = (float*)alloc((size_t)P * 192 * 4);
    float* X4   = (float*)alloc((size_t)P * 192 * 4);
    float* nllb = (float*)alloc((size_t)P * 4);
    float* stats = (float*)alloc((size_t)1154 * 4);   // S[2] + (sum,sq) x {c2,c3,c4}
    float* S    = stats;
    float* sum2 = stats + 2,   *sq2 = stats + 194;
    float* sum3 = stats + 386, *sq3 = stats + 578;
    float* sum4 = stats + 770, *sq4 = stats + 962;
    float* sc1 = (float*)alloc(96 * 4),  *sh1 = (float*)alloc(96 * 4);
    float* sc2 = (float*)alloc(192 * 4), *sh2 = (float*)alloc(192 * 4);
    float* sc3 = (float*)alloc(192 * 4), *sh3 = (float*)alloc(192 * 4);
    float* sc4 = (float*)alloc(192 * 4), *sh4 = (float*)alloc(192 * 4);
    float* muP = (float*)alloc(4);

    auto cdiv = [](int a, int b) { return (a + b - 1) / b; };

    zero_f<<<cdiv(1154, 256), 256, 0, stream>>>(stats, 1154);

    // pack weights (masked MADE + plain conv)
    pack_w<<<cdiv(1024 * 640,  256), 256, 0, stream>>>(W_in,  wbf_in,  1024, 627,  1024, 640,  1, 0);
    pack_w<<<cdiv(1024 * 1024, 256), 256, 0, stream>>>(W_h1,  wbf_h1,  1024, 1024, 1024, 1024, 1, 0);
    pack_w<<<cdiv(1024 * 1024, 256), 256, 0, stream>>>(W_h2,  wbf_h2,  1024, 1024, 1024, 1024, 1, 0);
    pack_w<<<cdiv(1280 * 1024, 256), 256, 0, stream>>>(W_out, wbf_out, 1254, 1024, 1280, 1024, 1, 1);
    pack_w<<<cdiv(192 * 96,  256), 256, 0, stream>>>(cw2, wbf_c2, 180, 90,  192, 96,  0, 0);
    pack_w<<<cdiv(192 * 192, 256), 256, 0, stream>>>(cw3, wbf_c3, 180, 180, 192, 192, 0, 0);
    pack_w<<<cdiv(192 * 192, 256), 256, 0, stream>>>(cw4, wbf_c4, 180, 180, 192, 192, 0, 0);
    pack_w<<<cdiv(304 * 192, 256), 256, 0, stream>>>(cw5, wbf_c5, 300, 180, 304, 192, 0, 0);

    // ---- MADE MLP (4 WMMA GEMMs) ----
    gemm_wmma<<<dim3(4, 16), 256, 0, stream>>>(inputs, 627, wbf_in, 640, 627, b_in, 1024,
        h1, 1024, 0, 1, 0, nullptr, nullptr, nullptr, nullptr, nullptr, 0, nullptr, nullptr);
    gemm_wmma<<<dim3(4, 16), 256, 0, stream>>>(h1, 1024, wbf_h1, 1024, 1024, b_h1, 1024,
        h2, 1024, 0, 1, 0, nullptr, nullptr, nullptr, nullptr, nullptr, 0, nullptr, nullptr);
    gemm_wmma<<<dim3(4, 16), 256, 0, stream>>>(h2, 1024, wbf_h2, 1024, 1024, b_h2, 1024,
        h3, 1024, 0, 0, 0, nullptr, nullptr, nullptr, nullptr, nullptr, 0, nullptr, nullptr);
    gemm_wmma<<<dim3(4, 20), 256, 0, stream>>>(h3, 1024, wbf_out, 1024, 1024, b_out, 1254,
        outm, 1280, 0, 0, 0, nullptr, nullptr, nullptr, nullptr, nullptr, 0, nullptr, nullptr);

    // ---- conv1 BN stats analytically from theta moments ----
    theta_stats<<<512, 256, 0, stream>>>(outm, 1280, S);
    bn1_fin<<<1, 128, 0, stream>>>(S, cw1, cb1, g1, be1, sc1, sh1);

    // ---- conv2 (A = virtual bn1(conv1(theta)), fused), + BN2 stats ----
    gemm_wmma<<<dim3(P / 128, 3), 256, 0, stream>>>(nullptr, 0, wbf_c2, 96, 90, cb2, 180,
        X2, 192, 2, 0, 1, sc1, sh1, cw1, cb1, outm, 1280, sum2, sq2);
    bn_fin<<<1, 256, 0, stream>>>(sum2, sq2, g2, be2, sc2, sh2, 180);

    // ---- conv3 (A = relu(bn2(X2)) fused), + BN3 stats ----
    gemm_wmma<<<dim3(P / 128, 3), 256, 0, stream>>>(X2, 192, wbf_c3, 192, 180, cb3, 180,
        X3, 192, 1, 0, 1, sc2, sh2, nullptr, nullptr, nullptr, 0, sum3, sq3);
    bn_fin<<<1, 256, 0, stream>>>(sum3, sq3, g3, be3, sc3, sh3, 180);

    // ---- conv4, + BN4 stats ----
    gemm_wmma<<<dim3(P / 128, 3), 256, 0, stream>>>(X3, 192, wbf_c4, 192, 180, cb4, 180,
        X4, 192, 1, 0, 1, sc3, sh3, nullptr, nullptr, nullptr, 0, sum4, sq4);
    bn_fin<<<1, 256, 0, stream>>>(sum4, sq4, g4, be4, sc4, sh4, 180);

    // ---- fused conv5 + log_softmax + NLL (pred never hits HBM) ----
    conv5_nll<<<P / 128, 256, 0, stream>>>(X4, wbf_c5, cb5, sc4, sh4, inputs, nllb);

    // ---- final loss ----
    nll_head_mean<<<1, 256, 0, stream>>>(nllb, muP);
    final_ll<<<512, 256, 0, stream>>>(outm, 1280, inputs, nllb, muP, (float*)d_out);
}
